// MusicalFeatureExtractor_64381559767345
// MI455X (gfx1250) — matrix operand order repaired and checked by reading, not compiled, not run on hardware
//
#include <hip/hip_runtime.h>

// ---------- types ----------
typedef __bf16 bf16_t;
typedef bf16_t v8bf   __attribute__((ext_vector_type(8)));
typedef bf16_t v16bf  __attribute__((ext_vector_type(16)));
typedef float  v8f    __attribute__((ext_vector_type(8)));
typedef float  v4f    __attribute__((ext_vector_type(4)));
typedef int    v4i    __attribute__((ext_vector_type(4)));
typedef unsigned short u16;

static __device__ __forceinline__ u16 f2bf(float f) {
  unsigned u = __float_as_uint(f);
  unsigned r = u + 0x7FFFu + ((u >> 16) & 1u);   // round-to-nearest-even
  return (u16)(r >> 16);
}

#define BATCH 32
#define SEQ   4096

// ---------- 0: zero scratch (16B stores) ----------
__global__ void __launch_bounds__(256) k_zero(v4i* __restrict__ p, long n) {
  long i = (long)blockIdx.x * 256 + threadIdx.x;
  if (i < n) { v4i z = {0, 0, 0, 0}; p[i] = z; }
}

// ---------- 1: plain f32 -> bf16 convert (fc_w) ----------
__global__ void __launch_bounds__(256) k_cvt(const float* __restrict__ s,
                                             u16* __restrict__ d, int n) {
  int i = blockIdx.x * 256 + threadIdx.x;
  if (i < n) d[i] = f2bf(s[i]);
}

// ---------- 2: conv weight transpose [co][ci][K] f32 -> [k][co][ci] bf16 ----------
__global__ void __launch_bounds__(256) k_wtr(const float* __restrict__ s,
                                             u16* __restrict__ d,
                                             int Cout, int Cin, int K, int total) {
  int i = blockIdx.x * 256 + threadIdx.x;
  if (i >= total) return;
  int k  = i / (Cout * Cin);
  int r  = i - k * Cout * Cin;
  int co = r / Cin;
  int ci = r - co * Cin;
  d[i] = f2bf(s[(co * Cin + ci) * K + k]);
}

// ---------- 3: one-hot-pitch conv (h1 / m1): gather instead of GEMM ----------
// out[b][pos+4][co] bf16, co=64, zero-padded rows
__global__ void __launch_bounds__(256) k_onehot_conv(const int* __restrict__ tokens,
                                                     const float* __restrict__ w,
                                                     const float* __restrict__ bias,
                                                     u16* __restrict__ out,
                                                     int K, int pad, int Lout, int rows) {
  int bpb = (Lout + 3) >> 2;                 // 4 positions per block
  int b   = blockIdx.x / bpb;
  int pos = (blockIdx.x % bpb) * 4 + (threadIdx.x >> 6);
  int co  = threadIdx.x & 63;
  if (pos >= Lout) return;
  const int* tb = tokens + b * SEQ;
  float acc = bias[co];
  for (int k = 0; k < K; k++) {
    int ip = pos + k - pad;
    if ((unsigned)ip < (unsigned)SEQ) {
      int tok = tb[ip];
      if (tok < 128) acc += w[(co * 128 + tok) * K + k];
    }
  }
  acc = fmaxf(acc, 0.f);
  out[((size_t)b * rows + pos + 4) * 64 + co] = f2bf(acc);
}

// ---------- 4: rhythm conv r1 (binary time-shift indicator), stride 2 ----------
__global__ void __launch_bounds__(256) k_r1(const int* __restrict__ tokens,
                                            const float* __restrict__ w,
                                            const float* __restrict__ bias,
                                            u16* __restrict__ out) {
  int b   = blockIdx.x >> 8;                 // 256 blocks per batch
  int pos = (blockIdx.x & 255) * 8 + (threadIdx.x >> 5);
  int co  = threadIdx.x & 31;
  const int* tb = tokens + b * SEQ;
  float acc = bias[co];
  for (int k = 0; k < 8; k++) {
    int ip = 2 * pos + k - 3;
    if ((unsigned)ip < (unsigned)SEQ) {
      int tok = tb[ip];
      if (tok >= 256 && tok < 768) acc += w[co * 8 + k];
    }
  }
  acc = fmaxf(acc, 0.f);
  out[((size_t)b * 2056 + pos + 4) * 32 + co] = f2bf(acc);
}

// ---------- 5: velocity conv d1 (6-wide one-hot gather), f32 out ----------
__global__ void __launch_bounds__(256) k_d1(const int* __restrict__ tokens,
                                            const float* __restrict__ w,
                                            const float* __restrict__ bias,
                                            float* __restrict__ out) {
  const int bpb = 257;                        // ceil(4097/16)
  int b   = blockIdx.x / bpb;
  int pos = (blockIdx.x % bpb) * 16 + (threadIdx.x >> 4);
  int c   = threadIdx.x & 15;
  if (pos >= 4097) return;
  const int* tb = tokens + b * SEQ;
  float acc = bias[c];
  for (int k = 0; k < 4; k++) {
    int ip = pos + k - 2;
    if ((unsigned)ip < (unsigned)SEQ) {
      int tok = tb[ip];
      if (tok >= 768) {
        int v = tok - 768; if (v > 5) v = 5;
        acc += w[(c * 6 + v) * 4 + k];
      }
    }
  }
  acc = fmaxf(acc, 0.f);
  out[((size_t)b * 4099 + pos + 1) * 16 + c] = acc;
}

// ---------- 6: WMMA im2col conv + relu + fused adaptive-pool accumulate ----------
// One wave = one 16(co) x 16(pos) tile; A = weights [k][co][ci] bf16,
// B = position-major activations [b][row][ci] bf16 (zero padded rows).
__global__ void __launch_bounds__(256)
k_wmma_conv(const u16* __restrict__ in_t, const u16* __restrict__ Wt,
            const float* __restrict__ bias, float* __restrict__ facc,
            int Cin, int Ktaps, int stride, int cpad, int Lout, int rows,
            int nseg, int Cout, int ntiles) {
  int lane = threadIdx.x & 31;
  int wave = threadIdx.x >> 5;
  int tpb  = (Cout >> 4) * ntiles;
  int tile = blockIdx.x * 8 + wave;
  if (tile >= BATCH * tpb) return;            // wave-uniform
  int b  = tile / tpb;
  int r  = tile - b * tpb;
  int mt = r / ntiles;
  int nt = r - mt * ntiles;
  int half = lane >> 4, ln = lane & 15;

  v8f acc = {};
  int p = stride * (nt * 16 + ln) - cpad;     // logical input pos for tap 0 (>= -4)
  const u16* bbase = in_t + ((size_t)b * rows + (p + 4)) * Cin + half * 16;
  const u16* abase = Wt + (size_t)(mt * 16 + ln) * Cin;

  for (int k = 0; k < Ktaps; k++) {
    const u16* arow = abase + (size_t)k * Cout * Cin;
    const u16* brow = bbase + (size_t)k * Cin; // row p+k
    __builtin_prefetch(brow + Cin, 0, 1);      // -> global_prefetch_b8
    for (int kc = 0; kc < Cin; kc += 32) {
      v16bf a = *(const v16bf*)(arow + kc + half * 16);
      v16bf bm = *(const v16bf*)(brow + kc);   // 16 contiguous K values (32B)
      acc = __builtin_amdgcn_wmma_f32_16x16x32_bf16(false, a, false, bm,
                                                    (short)0, acc, false, false);
    }
  }

  int seglen = Lout / nseg;
  int seg = (nt * 16) / seglen;
#pragma unroll
  for (int v = 0; v < 8; v++) {
    int co = mt * 16 + v + half * 8;
    float val = fmaxf(acc[v] + bias[co], 0.f);
    val += __shfl_xor(val, 1, 32);
    val += __shfl_xor(val, 2, 32);
    val += __shfl_xor(val, 4, 32);
    val += __shfl_xor(val, 8, 32);             // sum over 16 positions (half-wave)
    if (ln == 0) atomicAdd(&facc[(size_t)b * 512 + co * nseg + seg], val);
  }
}

// ---------- 7: d2 conv + faithful non-uniform double pooling ----------
__global__ void __launch_bounds__(256) k_d2(const float* __restrict__ d1buf,
                                            const float* __restrict__ w,
                                            const float* __restrict__ bias,
                                            float* __restrict__ facc) {
  __shared__ float pooled[2048];
  int b = blockIdx.x;
  const float* dbase = d1buf + (size_t)b * 4099 * 16;
  for (int idx = threadIdx.x; idx < 2048; idx += 256) {
    int c = idx >> 7, j = idx & 127;
    int start = (j * 4097) >> 7;
    int end   = ((j + 1) * 4097 + 127) >> 7;
    float sum = 0.f;
    for (int l = start; l < end; l++) {
      float v = bias[c];
      const float* r0 = dbase + (size_t)l * 16;   // rows l..l+2 == input idx l+k-1
      for (int k = 0; k < 3; k++)
        for (int ci = 0; ci < 16; ci++)
          v += w[(c * 16 + ci) * 3 + k] * r0[k * 16 + ci];
      sum += fmaxf(v, 0.f);
    }
    pooled[idx] = sum / (float)(end - start);
  }
  __syncthreads();
  if (threadIdx.x < 128) {
    int i = threadIdx.x, c = i >> 3, g = i & 7;
    float s = 0.f;
    for (int j = 0; j < 16; j++) s += pooled[c * 128 + g * 16 + j];
    facc[(size_t)b * 512 + 384 + i] = s * (1.f / 16.f);
  }
}

// ---------- 8: scale accumulators -> combined bf16 ----------
__global__ void __launch_bounds__(256) k_scale(const float* __restrict__ facc,
                                               u16* __restrict__ comb) {
  int idx = blockIdx.x * 256 + threadIdx.x;   // 32*512
  int i = idx & 511;
  int br = i >> 7;
  float sc = (br <= 1) ? (1.f / 512.f) : (br == 2 ? (1.f / 1024.f) : 1.f);
  comb[idx] = f2bf(facc[idx] * sc);
}

// ---------- 9: FC  feats[32,512] = combined @ fc_w^T + fc_b  (WMMA) ----------
__global__ void __launch_bounds__(256) k_fc(const u16* __restrict__ A,
                                            const u16* __restrict__ Bw,
                                            const float* __restrict__ bias,
                                            float* __restrict__ feats) {
  int lane = threadIdx.x & 31;
  int wave = threadIdx.x >> 5;
  int tile = blockIdx.x * 8 + wave;           // 64 tiles: 2 x 32
  int mt = tile >> 5, nt = tile & 31;
  int half = lane >> 4, ln = lane & 15;
  v8f acc = {};
  const u16* arow = A  + (size_t)(mt * 16 + ln) * 512;
  const u16* brow = Bw + (size_t)(nt * 16 + ln) * 512;   // fc_w row j contiguous in k
  for (int kb = 0; kb < 512; kb += 32) {
    v16bf a = *(const v16bf*)(arow + kb + half * 16);
    v16bf bm = *(const v16bf*)(brow + kb + half * 16);
    acc = __builtin_amdgcn_wmma_f32_16x16x32_bf16(false, a, false, bm,
                                                  (short)0, acc, false, false);
  }
  int j = nt * 16 + ln;
  float bj = bias[j];
#pragma unroll
  for (int v = 0; v < 8; v++) {
    int row = mt * 16 + v + half * 8;
    feats[(size_t)row * 512 + j] = acc[v] + bj;
  }
}

// ---------- 10: broadcast feats -> [B,S,512] with streaming (NT) stores ----------
__global__ void __launch_bounds__(256) k_broadcast(const float* __restrict__ feats,
                                                   float* __restrict__ out) {
  __shared__ float row[512];
  int b = blockIdx.x >> 7;                    // 32 batches x 128 chunks of 32 rows
  int chunk = blockIdx.x & 127;
  row[threadIdx.x]       = feats[b * 512 + threadIdx.x];
  row[threadIdx.x + 256] = feats[b * 512 + threadIdx.x + 256];
  __syncthreads();
  int tcol = (threadIdx.x & 127) * 4;
  int trow = threadIdx.x >> 7;                // 0..1
  v4f v = *(const v4f*)&row[tcol];
  size_t base = ((size_t)b * SEQ + (size_t)chunk * 32) * 512;
  for (int s = trow; s < 32; s += 2)
    __builtin_nontemporal_store(v, (v4f*)(out + base + (size_t)s * 512 + tcol));
}

// ---------- host launch ----------
extern "C" void kernel_launch(void* const* d_in, const int* in_sizes, int n_in,
                              void* d_out, int out_size, void* d_ws, size_t ws_size,
                              hipStream_t stream) {
  (void)in_sizes; (void)n_in; (void)out_size; (void)ws_size;
  const int*   tokens = (const int*)  d_in[0];
  const float* r1w = (const float*)d_in[1];  const float* r1b = (const float*)d_in[2];
  const float* r2w = (const float*)d_in[3];  const float* r2b = (const float*)d_in[4];
  const float* h1w = (const float*)d_in[5];  const float* h1b = (const float*)d_in[6];
  const float* h2w = (const float*)d_in[7];  const float* h2b = (const float*)d_in[8];
  const float* m1w = (const float*)d_in[9];  const float* m1b = (const float*)d_in[10];
  const float* m2w = (const float*)d_in[11]; const float* m2b = (const float*)d_in[12];
  const float* d1w = (const float*)d_in[13]; const float* d1b = (const float*)d_in[14];
  const float* d2w = (const float*)d_in[15]; const float* d2b = (const float*)d_in[16];
  const float* fcw = (const float*)d_in[17]; const float* fcb = (const float*)d_in[18];

  char* ws = (char*)d_ws;
  // zero-initialized region (padded bf16 activations + f32 d1 + accumulators)
  const size_t o_h1t  = 0;                         // 32 x 4105 x 64 bf16
  const size_t o_m1t  = o_h1t + 16814080;          // 32 x 4104 x 64 bf16
  const size_t o_r1t  = o_m1t + 16809984;          // 32 x 2056 x 32 bf16
  const size_t o_d1   = o_r1t + 4210688;           // 32 x 4099 x 16 f32
  const size_t o_facc = o_d1  + 8394752;           // 32 x 512 f32
  const size_t zb     = o_facc + 65536;            // 46,295,040 B
  // non-zeroed region
  const size_t o_fcw  = zb;                        // 512x512 bf16
  const size_t o_wh2  = o_fcw + 524288;            // 8x32x64 bf16
  const size_t o_wm2  = o_wh2 + 32768;             // 3x32x64 bf16
  const size_t o_wr2  = o_wm2 + 12288;             // 4x64x32 bf16
  const size_t o_comb = o_wr2 + 16384;             // 32x512 bf16
  const size_t o_feat = o_comb + 32768;            // 32x512 f32

  u16*   h1t  = (u16*)(ws + o_h1t);
  u16*   m1t  = (u16*)(ws + o_m1t);
  u16*   r1t  = (u16*)(ws + o_r1t);
  float* d1f  = (float*)(ws + o_d1);
  float* facc = (float*)(ws + o_facc);
  u16*   fcwb = (u16*)(ws + o_fcw);
  u16*   wh2  = (u16*)(ws + o_wh2);
  u16*   wm2  = (u16*)(ws + o_wm2);
  u16*   wr2  = (u16*)(ws + o_wr2);
  u16*   comb = (u16*)(ws + o_comb);
  float* feat = (float*)(ws + o_feat);
  float* out  = (float*)d_out;

  // 0) zero scratch
  long n16 = (long)(zb / 16);
  k_zero<<<dim3((unsigned)((n16 + 255) / 256)), 256, 0, stream>>>((v4i*)ws, n16);
  // 1) weight prep (bf16 / transposed)
  k_cvt<<<1024, 256, 0, stream>>>(fcw, fcwb, 512 * 512);
  k_wtr<<<64, 256, 0, stream>>>(h2w, wh2, 32, 64, 8, 16384);
  k_wtr<<<24, 256, 0, stream>>>(m2w, wm2, 32, 64, 3, 6144);
  k_wtr<<<32, 256, 0, stream>>>(r2w, wr2, 64, 32, 4, 8192);
  // 2) sparse first convs -> padded, position-major bf16
  k_onehot_conv<<<32 * 1025, 256, 0, stream>>>(tokens, h1w, h1b, h1t, 12, 6, 4097, 4105);
  k_onehot_conv<<<32 * 1024, 256, 0, stream>>>(tokens, m1w, m1b, m1t, 5, 2, 4096, 4104);
  k_r1<<<32 * 256, 256, 0, stream>>>(tokens, r1w, r1b, r1t);
  k_d1<<<32 * 257, 256, 0, stream>>>(tokens, d1w, d1b, d1f);
  // 3) WMMA second convs with fused relu + pooled accumulation into facc
  k_wmma_conv<<<1024, 256, 0, stream>>>(r1t, wr2, r2b, facc + 0,   32, 4, 2, 1, 1024, 2056, 2, 64, 64);
  k_wmma_conv<<<1024, 256, 0, stream>>>(h1t, wh2, h2b, facc + 128, 64, 8, 2, 3, 2048, 4105, 4, 32, 128);
  k_wmma_conv<<<2048, 256, 0, stream>>>(m1t, wm2, m2b, facc + 256, 64, 3, 1, 1, 4096, 4104, 4, 32, 256);
  k_d2<<<32, 256, 0, stream>>>(d1f, d2w, d2b, facc);
  // 4) finalize combined, FC via WMMA, broadcast with NT stores
  k_scale<<<64, 256, 0, stream>>>(facc, comb);
  k_fc<<<8, 256, 0, stream>>>(comb, fcwb, fcb, feat);
  k_broadcast<<<4096, 256, 0, stream>>>(feat, out);
}
